// MambaClassifier_49898930045043
// MI455X (gfx1250) — matrix operand order repaired and checked by reading, not compiled, not run on hardware
//
#include <hip/hip_runtime.h>
#include <hip/hip_bf16.h>

// ---------------- model dims (must match reference) ----------------
#define BATCH   4
#define SEQ     1024
#define DMODEL  1024
#define DINNER  2048
#define DSTATE  16
#define DCONV   4
#define DTRANK  64
#define NCLS    10
#define NTOK    (BATCH * SEQ)                 // 4096 tokens
#define XPROJ_E (DTRANK + 2 * DSTATE)         // 96
#define NCHUNK  16                            // scan chunks
#define CLEN    (SEQ / NCHUNK)                // 64 steps per chunk
#define KCHUNK  64                            // GEMM K staging chunk

typedef __attribute__((ext_vector_type(16))) __bf16 v16bf;
typedef __attribute__((ext_vector_type(8)))  __bf16 v8bf;
typedef __attribute__((ext_vector_type(8)))  float  v8f;

// ---------------- fp32 -> (hi,lo) bf16 split ----------------
__global__ __launch_bounds__(256)
void split_bf16_kernel(const float* __restrict__ src,
                       __bf16* __restrict__ hi, __bf16* __restrict__ lo, int n) {
    int i = blockIdx.x * blockDim.x + threadIdx.x;
    if (i >= n) return;
    float x = src[i];
    __bf16 h = (__bf16)x;
    hi[i] = h;
    lo[i] = (__bf16)(x - (float)h);
}

// ---------------- LayerNorm (one block per token, dim=1024) ----------------
__global__ __launch_bounds__(256)
void layernorm_kernel(const float* __restrict__ x,
                      const float* __restrict__ w, const float* __restrict__ b,
                      __bf16* __restrict__ hi, __bf16* __restrict__ lo,
                      float* __restrict__ outf, int dim) {
    int t = blockIdx.x;
    const float* row = x + (size_t)t * dim;
    __shared__ float red0[8], red1[8];
    float s = 0.f, s2 = 0.f;
    for (int i = threadIdx.x; i < dim; i += blockDim.x) {
        float v = row[i]; s += v; s2 += v * v;
    }
    for (int off = 16; off > 0; off >>= 1) {
        s  += __shfl_xor(s,  off);
        s2 += __shfl_xor(s2, off);
    }
    int wid = threadIdx.x >> 5;
    if ((threadIdx.x & 31) == 0) { red0[wid] = s; red1[wid] = s2; }
    __syncthreads();
    if (threadIdx.x == 0) {
        float a = 0.f, c = 0.f;
        int nw = blockDim.x >> 5;
        for (int i = 0; i < nw; i++) { a += red0[i]; c += red1[i]; }
        red0[0] = a; red1[0] = c;
    }
    __syncthreads();
    float mean = red0[0] / dim;
    float var  = red1[0] / dim - mean * mean;
    float inv  = rsqrtf(var + 1e-5f);
    for (int i = threadIdx.x; i < dim; i += blockDim.x) {
        float v = (row[i] - mean) * inv * w[i] + b[i];
        if (outf) outf[(size_t)t * dim + i] = v;
        if (hi) {
            __bf16 h = (__bf16)v;
            hi[(size_t)t * dim + i] = h;
            lo[(size_t)t * dim + i] = (__bf16)(v - (float)h);
        }
    }
}

// ---------------- split-bf16 WMMA TN GEMM with async-LDS staging ----------------
// Block = 256 threads = 8 waves (2 mt x 4 nt) computing a 32x64 f32 tile of
// C[M,N] (+)= A[M,K] * B[N,K]^T.  A/B are bf16 hi/lo pairs.  Each K-chunk of 64
// (A: 2x32x64, B: 2x64x64 bf16 = 24KB) is staged global->LDS with
// GLOBAL_LOAD_ASYNC_TO_LDS_B128 (ASYNCcnt), double buffered so the copy of
// chunk i+1 overlaps the 6 v_wmma of chunk i.
__global__ __launch_bounds__(256)
void wmma_gemm_tn_lds(const __bf16* __restrict__ Ah, const __bf16* __restrict__ Al,
                      const __bf16* __restrict__ Bh, const __bf16* __restrict__ Bl,
                      float* __restrict__ C, int M, int N, int K, int addC) {
    __shared__ __align__(128) __bf16 sA[2][2][32][KCHUNK];   // [buf][hi/lo][row][k] 16KB
    __shared__ __align__(128) __bf16 sB[2][2][64][KCHUNK];   // 32KB

    int tid  = threadIdx.x;
    int wave = tid >> 5;
    int lane = tid & 31;
    int ntb  = N >> 6;                       // N blocks of 64
    int mb   = blockIdx.x / ntb;
    int nb   = blockIdx.x % ntb;
    int m0   = mb * 32, n0 = nb * 64;

    // stage one K-chunk: 1536 x 16B pieces, 6 per thread, via async LDS loads
    auto issue_chunk = [&](int buf, int kc) {
        #pragma unroll
        for (int j = 0; j < 6; j++) {
            int p = tid + j * 256;
            const __bf16* src;
            unsigned dst;
            if (p < 512) {                   // A: hl(2) x row(32) x piece(8)
                int hl = p >> 8, rem = p & 255, row = rem >> 3, pc = rem & 7;
                src = (hl ? Al : Ah) + (size_t)(m0 + row) * K + kc + pc * 8;
                dst = (unsigned)(uintptr_t)(&sA[buf][hl][row][pc * 8]);
            } else {                         // B: hl(2) x row(64) x piece(8)
                int q = p - 512;
                int hl = q >> 9, rem = q & 511, row = rem >> 3, pc = rem & 7;
                src = (hl ? Bl : Bh) + (size_t)(n0 + row) * K + kc + pc * 8;
                dst = (unsigned)(uintptr_t)(&sB[buf][hl][row][pc * 8]);
            }
            unsigned long long ga = (unsigned long long)(uintptr_t)src;
            asm volatile("global_load_async_to_lds_b128 %0, %1, off"
                         :: "v"(dst), "v"(ga) : "memory");
        }
    };

    int mtl  = wave >> 2;                    // 0..1
    int ntl  = wave & 3;                     // 0..3
    int r    = lane & 15;
    int half = lane >> 4;                    // 0 or 1

    v8f acc = {};
    issue_chunk(0, 0);
    int nchunks = K / KCHUNK;
    for (int ci = 0; ci < nchunks; ci++) {
        if (ci + 1 < nchunks) {
            issue_chunk((ci + 1) & 1, (ci + 1) * KCHUNK);
            asm volatile("s_wait_asynccnt 0x6" ::: "memory");  // oldest 6 (this chunk) done
        } else {
            asm volatile("s_wait_asynccnt 0x0" ::: "memory");
        }
        __syncthreads();                     // all waves' stages visible
        int buf = ci & 1;
        #pragma unroll
        for (int ks = 0; ks < KCHUNK; ks += 32) {
            // A per-lane k order (this code's contiguous order, same as B): lanes0-15 K0..15, lanes16-31 K16..31
            const __bf16* aH = &sA[buf][0][mtl * 16 + r][ks + half * 16];
            const __bf16* aL = &sA[buf][1][mtl * 16 + r][ks + half * 16];
            // B per-lane k order (this code's contiguous order, same as A): lanes0-15 K0..15, lanes16-31 K16..31
            const __bf16* bH = &sB[buf][0][ntl * 16 + r][ks + half * 16];
            const __bf16* bL = &sB[buf][1][ntl * 16 + r][ks + half * 16];
            v8bf a0h = *(const v8bf*)(aH);
            v8bf a1h = *(const v8bf*)(aH + 8);
            v8bf a0l = *(const v8bf*)(aL);
            v8bf a1l = *(const v8bf*)(aL + 8);
            v16bf avh = __builtin_shufflevector(a0h, a1h, 0,1,2,3,4,5,6,7,8,9,10,11,12,13,14,15);
            v16bf avl = __builtin_shufflevector(a0l, a1l, 0,1,2,3,4,5,6,7,8,9,10,11,12,13,14,15);
            v16bf bvh = *(const v16bf*)(bH);
            v16bf bvl = *(const v16bf*)(bL);
            acc = __builtin_amdgcn_wmma_f32_16x16x32_bf16(false, avh, false, bvh, (short)0, acc, false, false);
            acc = __builtin_amdgcn_wmma_f32_16x16x32_bf16(false, avh, false, bvl, (short)0, acc, false, false);
            acc = __builtin_amdgcn_wmma_f32_16x16x32_bf16(false, avl, false, bvh, (short)0, acc, false, false);
        }
        __syncthreads();                     // done reading buf before it is overwritten
    }

    // C layout: VGPR j -> row +j (+8 for lanes>=16), col = lane&15
    int crow = m0 + mtl * 16 + half * 8;
    int ccol = n0 + ntl * 16 + r;
    float* cp = C + (size_t)crow * N + ccol;
    if (addC) {
        #pragma unroll
        for (int j = 0; j < 8; j++) cp[(size_t)j * N] += acc[j];
    } else {
        #pragma unroll
        for (int j = 0; j < 8; j++) cp[(size_t)j * N] = acc[j];
    }
}

// ---------------- depthwise causal conv (k=4) + SiLU ----------------
__global__ __launch_bounds__(256)
void conv_silu_kernel(const float* __restrict__ xz, const float* __restrict__ cw,
                      const float* __restrict__ cb, float* __restrict__ xc) {
    int i = blockIdx.x * blockDim.x + threadIdx.x;
    if (i >= NTOK * DINNER) return;
    int c = i % DINNER;
    int t = (i / DINNER) % SEQ;
    int b = i / (DINNER * SEQ);
    float acc = cb[c];
    #pragma unroll
    for (int k = 0; k < DCONV; k++) {
        int tt = t + k - (DCONV - 1);
        if (tt >= 0)
            acc += cw[c * DCONV + k] * xz[((size_t)(b * SEQ + tt)) * (2 * DINNER) + c];
    }
    xc[i] = acc / (1.f + __expf(-acc));     // silu
}

// ---------------- x_proj: [NTOK,2048] x [96,2048]^T (fp32 VALU, exact) ----------------
__global__ __launch_bounds__(256)
void xproj_kernel(const float* __restrict__ xc, const float* __restrict__ w,
                  float* __restrict__ xdbl) {
    int i = blockIdx.x * blockDim.x + threadIdx.x;
    if (i >= NTOK * XPROJ_E) return;
    int e = i % XPROJ_E;
    int tok = i / XPROJ_E;
    const float* xr = xc + (size_t)tok * DINNER;
    const float* wr = w + (size_t)e * DINNER;
    float acc = 0.f;
    #pragma unroll 4
    for (int k = 0; k < DINNER; k++) acc += xr[k] * wr[k];
    xdbl[i] = acc;
}

// ---------------- dt = softplus(dt_r @ dtp_w^T + dtp_b) ----------------
__global__ __launch_bounds__(256)
void dt_kernel(const float* __restrict__ xdbl, const float* __restrict__ dtw,
               const float* __restrict__ dtb, float* __restrict__ dt) {
    int i = blockIdx.x * blockDim.x + threadIdx.x;
    if (i >= NTOK * DINNER) return;
    int c = i % DINNER;
    int tok = i / DINNER;
    const float* r = xdbl + (size_t)tok * XPROJ_E;   // first DTRANK entries = dt_r
    const float* w = dtw + (size_t)c * DTRANK;
    float acc = dtb[c];
    #pragma unroll 8
    for (int k = 0; k < DTRANK; k++) acc += r[k] * w[k];
    dt[i] = (acc > 20.f) ? acc : log1pf(__expf(acc));
}

// ---------------- chunked selective scan ----------------
// h_t = dA_t*h_{t-1} + u_t is associative: split L into NCHUNK chunks.
// Pass A: per (b,chunk,c) run chunk from h=0; emit decay product P[n] and
//         endpoint H[n].              (131072 threads)
// Pass B: per (b,c,n) 16-step cross-chunk prefix -> true initial state.
// Pass C: replay chunk from true init; fuse +x*D, silu(z) gate, bf16 split.
__global__ __launch_bounds__(256)
void scan_chunk_kernel(const float* __restrict__ dt, const float* __restrict__ xc,
                       const float* __restrict__ xdbl, const float* __restrict__ A_log,
                       float* __restrict__ chunkP, float* __restrict__ chunkH) {
    int i = blockIdx.x * blockDim.x + threadIdx.x;       // B*NCHUNK*DINNER
    if (i >= BATCH * NCHUNK * DINNER) return;
    int c  = i % DINNER;
    int ch = (i / DINNER) % NCHUNK;
    int b  = i / (DINNER * NCHUNK);
    float Aa[DSTATE], h[DSTATE], P[DSTATE];
    #pragma unroll
    for (int n = 0; n < DSTATE; n++) {
        Aa[n] = -__expf(A_log[(size_t)c * DSTATE + n]);
        h[n] = 0.f; P[n] = 1.f;
    }
    for (int t = ch * CLEN; t < (ch + 1) * CLEN; t++) {
        size_t tok = (size_t)b * SEQ + t;
        float dtv = dt[tok * DINNER + c];
        float xv  = xc[tok * DINNER + c];
        const float* bc = xdbl + tok * XPROJ_E + DTRANK;
        float dtx = dtv * xv;
        #pragma unroll
        for (int n = 0; n < DSTATE; n++) {
            float dA = __expf(dtv * Aa[n]);
            h[n] = dA * h[n] + dtx * bc[n];
            P[n] *= dA;
        }
    }
    size_t o = ((size_t)((b * NCHUNK + ch) * DINNER) + c) * DSTATE;
    #pragma unroll
    for (int n = 0; n < DSTATE; n++) { chunkP[o + n] = P[n]; chunkH[o + n] = h[n]; }
}

__global__ __launch_bounds__(256)
void scan_prefix_kernel(const float* __restrict__ chunkP, const float* __restrict__ chunkH,
                        float* __restrict__ initH) {
    int i = blockIdx.x * blockDim.x + threadIdx.x;       // B*DINNER*DSTATE
    if (i >= BATCH * DINNER * DSTATE) return;
    int n = i % DSTATE;
    int c = (i / DSTATE) % DINNER;
    int b = i / (DSTATE * DINNER);
    float h = 0.f;
    for (int ch = 0; ch < NCHUNK; ch++) {
        size_t o = ((size_t)((b * NCHUNK + ch) * DINNER) + c) * DSTATE + n;
        initH[o] = h;
        h = chunkP[o] * h + chunkH[o];
    }
}

__global__ __launch_bounds__(256)
void scan_final_kernel(const float* __restrict__ dt, const float* __restrict__ xc,
                       const float* __restrict__ xdbl, const float* __restrict__ xz,
                       const float* __restrict__ A_log, const float* __restrict__ Dp,
                       const float* __restrict__ initH,
                       __bf16* __restrict__ yh, __bf16* __restrict__ yl) {
    int i = blockIdx.x * blockDim.x + threadIdx.x;       // B*NCHUNK*DINNER
    if (i >= BATCH * NCHUNK * DINNER) return;
    int c  = i % DINNER;
    int ch = (i / DINNER) % NCHUNK;
    int b  = i / (DINNER * NCHUNK);
    float Aa[DSTATE], h[DSTATE];
    size_t o0 = ((size_t)((b * NCHUNK + ch) * DINNER) + c) * DSTATE;
    #pragma unroll
    for (int n = 0; n < DSTATE; n++) {
        Aa[n] = -__expf(A_log[(size_t)c * DSTATE + n]);
        h[n] = initH[o0 + n];
    }
    float Dv = Dp[c];
    for (int t = ch * CLEN; t < (ch + 1) * CLEN; t++) {
        size_t tok = (size_t)b * SEQ + t;
        float dtv = dt[tok * DINNER + c];
        float xv  = xc[tok * DINNER + c];
        const float* bc = xdbl + tok * XPROJ_E + DTRANK;   // B[0:16], C[16:32]
        float dtx = dtv * xv;
        float y = 0.f;
        #pragma unroll
        for (int n = 0; n < DSTATE; n++) {
            float dA = __expf(dtv * Aa[n]);
            h[n] = dA * h[n] + dtx * bc[n];
            y += h[n] * bc[DSTATE + n];
        }
        y += xv * Dv;
        float z = xz[tok * (2 * DINNER) + DINNER + c];
        y *= z / (1.f + __expf(-z));                        // * silu(z)
        __bf16 hh = (__bf16)y;
        yh[tok * DINNER + c] = hh;
        yl[tok * DINNER + c] = (__bf16)(y - (float)hh);
    }
}

// ---------------- mean pool over sequence ----------------
__global__ __launch_bounds__(256)
void pool_kernel(const float* __restrict__ xn, float* __restrict__ pooled) {
    int i = blockIdx.x * blockDim.x + threadIdx.x;
    if (i >= BATCH * DMODEL) return;
    int d = i % DMODEL;
    int b = i / DMODEL;
    float s = 0.f;
    for (int t = 0; t < SEQ; t++) s += xn[((size_t)b * SEQ + t) * DMODEL + d];
    pooled[i] = s * (1.f / SEQ);
}

// ---------------- classifier head (tiny) ----------------
__global__ void head_kernel(const float* __restrict__ pooled,
                            const float* __restrict__ hw, const float* __restrict__ hb,
                            float* __restrict__ out) {
    int i = threadIdx.x;
    if (i >= BATCH * NCLS) return;
    int j = i % NCLS;
    int b = i / NCLS;
    const float* p = pooled + (size_t)b * DMODEL;
    const float* w = hw + (size_t)j * DMODEL;
    float acc = hb[j];
    for (int k = 0; k < DMODEL; k++) acc += p[k] * w[k];
    out[i] = acc;
}

// ======================= host side =======================
extern "C" void kernel_launch(void* const* d_in, const int* in_sizes, int n_in,
                              void* d_out, int out_size, void* d_ws, size_t ws_size,
                              hipStream_t stream) {
    const float* x      = (const float*)d_in[0];
    const float* ln1_w  = (const float*)d_in[1];
    const float* ln1_b  = (const float*)d_in[2];
    const float* in_w   = (const float*)d_in[3];
    const float* conv_w = (const float*)d_in[4];
    const float* conv_b = (const float*)d_in[5];
    const float* xp_w   = (const float*)d_in[6];
    const float* dtp_w  = (const float*)d_in[7];
    const float* dtp_b  = (const float*)d_in[8];
    const float* A_log  = (const float*)d_in[9];
    const float* D_par  = (const float*)d_in[10];
    const float* out_w  = (const float*)d_in[11];
    const float* norm_w = (const float*)d_in[12];
    const float* norm_b = (const float*)d_in[13];
    const float* head_w = (const float*)d_in[14];
    const float* head_b = (const float*)d_in[15];
    float* out = (float*)d_out;

    // ---- workspace carve (256B aligned) ----
    char* base = (char*)d_ws;
    size_t off = 0;
    auto alloc = [&](size_t bytes) -> void* {
        void* r = base + off;
        off = (off + bytes + 255) & ~(size_t)255;
        return r;
    };
    float*  x_cur  = (float*) alloc((size_t)NTOK * DMODEL * 4);
    float*  xz     = (float*) alloc((size_t)NTOK * 2 * DINNER * 4);
    float*  xc     = (float*) alloc((size_t)NTOK * DINNER * 4);     // reused as xn at the end
    float*  dt     = (float*) alloc((size_t)NTOK * DINNER * 4);
    float*  xdbl   = (float*) alloc((size_t)NTOK * XPROJ_E * 4);
    float*  pooled = (float*) alloc((size_t)BATCH * DMODEL * 4);
    float*  chunkP = (float*) alloc((size_t)BATCH * NCHUNK * DINNER * DSTATE * 4);
    float*  chunkH = (float*) alloc((size_t)BATCH * NCHUNK * DINNER * DSTATE * 4);
    float*  initH  = (float*) alloc((size_t)BATCH * NCHUNK * DINNER * DSTATE * 4);
    __bf16* h_hi   = (__bf16*)alloc((size_t)NTOK * DMODEL * 2);
    __bf16* h_lo   = (__bf16*)alloc((size_t)NTOK * DMODEL * 2);
    __bf16* y_hi   = (__bf16*)alloc((size_t)NTOK * DINNER * 2);
    __bf16* y_lo   = (__bf16*)alloc((size_t)NTOK * DINNER * 2);
    __bf16* w_hi   = (__bf16*)alloc((size_t)2 * DINNER * DMODEL * 2);
    __bf16* w_lo   = (__bf16*)alloc((size_t)2 * DINNER * DMODEL * 2);
    (void)ws_size; (void)in_sizes; (void)n_in; (void)out_size;

    hipMemcpyAsync(x_cur, x, (size_t)NTOK * DMODEL * 4,
                   hipMemcpyDeviceToDevice, stream);

    for (int l = 0; l < 2; l++) {
        const float* lw  = ln1_w  + (size_t)l * DMODEL;
        const float* lb  = ln1_b  + (size_t)l * DMODEL;
        const float* iw  = in_w   + (size_t)l * 2 * DINNER * DMODEL;
        const float* cw  = conv_w + (size_t)l * DINNER * DCONV;
        const float* cb  = conv_b + (size_t)l * DINNER;
        const float* xpw = xp_w   + (size_t)l * XPROJ_E * DINNER;
        const float* dtw = dtp_w  + (size_t)l * DINNER * DTRANK;
        const float* dtb = dtp_b  + (size_t)l * DINNER;
        const float* al  = A_log  + (size_t)l * DINNER * DSTATE;
        const float* dp  = D_par  + (size_t)l * DINNER;
        const float* ow  = out_w  + (size_t)l * DMODEL * DINNER;

        // 1. LayerNorm -> bf16 hi/lo
        layernorm_kernel<<<NTOK, 256, 0, stream>>>(x_cur, lw, lb, h_hi, h_lo, nullptr, DMODEL);

        // 2. in_proj GEMM: xz[4096,4096] = h[4096,1024] @ in_w[4096,1024]^T
        {
            int n = 2 * DINNER * DMODEL;
            split_bf16_kernel<<<(n + 255) / 256, 256, 0, stream>>>(iw, w_hi, w_lo, n);
            int blocks = (NTOK / 32) * ((2 * DINNER) / 64);      // 8192
            wmma_gemm_tn_lds<<<blocks, 256, 0, stream>>>(h_hi, h_lo, w_hi, w_lo,
                                                         xz, NTOK, 2 * DINNER, DMODEL, 0);
        }

        // 3. causal depthwise conv + SiLU
        conv_silu_kernel<<<(NTOK * DINNER + 255) / 256, 256, 0, stream>>>(xz, cw, cb, xc);

        // 4. x_proj (fp32 VALU, exact)
        xproj_kernel<<<(NTOK * XPROJ_E + 255) / 256, 256, 0, stream>>>(xc, xpw, xdbl);

        // 5. dt projection + softplus
        dt_kernel<<<(NTOK * DINNER + 255) / 256, 256, 0, stream>>>(xdbl, dtw, dtb, dt);

        // 6. chunked selective scan (3 passes)
        scan_chunk_kernel <<<(BATCH * NCHUNK * DINNER) / 256, 256, 0, stream>>>(dt, xc, xdbl, al, chunkP, chunkH);
        scan_prefix_kernel<<<(BATCH * DINNER * DSTATE) / 256, 256, 0, stream>>>(chunkP, chunkH, initH);
        scan_final_kernel <<<(BATCH * NCHUNK * DINNER) / 256, 256, 0, stream>>>(dt, xc, xdbl, xz, al, dp,
                                                                                initH, y_hi, y_lo);

        // 7. out_proj GEMM with residual: x_cur += y[4096,2048] @ out_w[1024,2048]^T
        {
            int n = DMODEL * DINNER;
            split_bf16_kernel<<<(n + 255) / 256, 256, 0, stream>>>(ow, w_hi, w_lo, n);
            int blocks = (NTOK / 32) * (DMODEL / 64);            // 2048
            wmma_gemm_tn_lds<<<blocks, 256, 0, stream>>>(y_hi, y_lo, w_hi, w_lo,
                                                         x_cur, NTOK, DMODEL, DINNER, 1);
        }
    }

    // final LayerNorm (fp32 out, reuse xc buffer), mean pool, head
    layernorm_kernel<<<NTOK, 256, 0, stream>>>(x_cur, norm_w, norm_b,
                                               nullptr, nullptr, xc, DMODEL);
    pool_kernel<<<(BATCH * DMODEL + 255) / 256, 256, 0, stream>>>(xc, pooled);
    head_kernel<<<1, 64, 0, stream>>>(pooled, head_w, head_b, out);
}
